// SDTPair_89739046682765
// MI455X (gfx1250) — matrix operand order repaired and checked by reading, not compiled, not run on hardware
//
#include <hip/hip_runtime.h>

// ---------------------------------------------------------------- types
typedef __bf16 bf16;
typedef bf16  v16bf __attribute__((ext_vector_type(16)));
typedef float v8f   __attribute__((ext_vector_type(8)));

#define BVAL 4
#define TVAL 1024
#define DVAL 1024
#define HVAL 16
#define DHV  64
#define FFNV 4096
#define KSEL 256

__device__ __forceinline__ bf16 f2bf(float f) {
  union { float f; unsigned u; } a; a.f = f;
  unsigned r = a.u + 0x7FFFu + ((a.u >> 16) & 1u);   // round-to-nearest-even
  union { unsigned short s; bf16 b; } o; o.s = (unsigned short)(r >> 16);
  return o.b;
}

// A-fragment K index for element i of the 16-element lane payload
// (contiguous: lane<16 holds K{0..15}, lane>=16 holds K{16..31}, matching B)
__device__ __forceinline__ int kidxA(int i, int half) {
  return i + 16 * half;
}

__device__ __forceinline__ v8f wmma_bf16(v16bf a, v16bf b, v8f c) {
  return __builtin_amdgcn_wmma_f32_16x16x32_bf16(false, a, false, b,
                                                 (short)0, c, false, false);
}

__device__ __forceinline__ v8f vzero8() {
  v8f z; for (int i = 0; i < 8; ++i) z[i] = 0.f; return z;
}

// ------------------------------------------------- transpose-convert weights
// src fp32 [Kd, Nd] row-major  ->  dst bf16 [Nd, Kd] row-major (one-time)
__global__ __launch_bounds__(256)
void k_cvt_t(const float* __restrict__ src, bf16* __restrict__ dst,
             int Kd, int Nd) {
  __shared__ bf16 t[32][33];
  const int bx = blockIdx.x * 32;          // n tile
  const int by = blockIdx.y * 32;          // k tile
  const int tx = threadIdx.x & 31, ty = threadIdx.x >> 5;   // 32x8
  for (int i = 0; i < 32; i += 8)
    t[ty + i][tx] = f2bf(src[(size_t)(by + ty + i) * Nd + bx + tx]);
  __syncthreads();
  for (int i = 0; i < 32; i += 8)
    dst[(size_t)(bx + ty + i) * Kd + by + tx] = t[tx][ty + i];
}

// ---------------------------------------------------------------- GEMM
// C[M,N] = A[M,K] @ B[K,N], A bf16 [M,K], B given TRANSPOSED bf16 [N,K].
// mode 0: Cf = AB               (fp32 out)
// mode 1: Cf = AB + aux         (fp32 out, residual add)
// mode 2: Cb = silu(aux) * AB   (bf16 out, SwiGLU fusion)
#define BM 128
#define BN 128
#define BKK 32
#define TST 40   // LDS tile row stride (elements): 80B rows, 16B aligned

__global__ __launch_bounds__(256)
void k_gemm(const bf16* __restrict__ A, const bf16* __restrict__ Bt,
            float* __restrict__ Cf, bf16* __restrict__ Cb,
            const float* __restrict__ aux,
            int M, int N, int K, int mode) {
  __shared__ bf16 As[2][BM][TST];   // ping-pong double buffer
  __shared__ bf16 Bs[2][BN][TST];   // [n][k]

  const int tid  = threadIdx.x;
  const int lane = tid & 31, w = tid >> 5;
  const int half = lane >> 4, l = lane & 15;
  const int wr = w >> 1, wc = w & 1;                 // 4x2 wave grid
  const int bm = blockIdx.y * BM, bn = blockIdx.x * BN;

  v8f acc[2][4];
  for (int mt = 0; mt < 2; ++mt)
    for (int nt = 0; nt < 4; ++nt) acc[mt][nt] = vzero8();

  // 512 chunks of 8 bf16 per 128x32 tile; this thread owns chunks tid, tid+256
  // stage = load + immediately store into LDS buffer `buf`
  // (nothing stays live across the WMMA burst -> no spills possible)
  auto stage = [&](int kt, int buf) {
    uint4 la[2], lb[2];
#pragma unroll
    for (int p = 0; p < 2; ++p) {
      const int c = tid + p * 256;
      const int r = c >> 2, kc = (c & 3) * 8;
      la[p] = *(const uint4*)(A  + (size_t)(bm + r) * K + kt + kc);
      lb[p] = *(const uint4*)(Bt + (size_t)(bn + r) * K + kt + kc);
    }
    if (kt + BKK < K) {  // prefetch one more slab ahead
      __builtin_prefetch(A  + (size_t)(bm + (tid >> 2)) * K + kt + BKK + (tid & 3) * 8, 0, 3);
      __builtin_prefetch(Bt + (size_t)(bn + (tid >> 2)) * K + kt + BKK + (tid & 3) * 8, 0, 3);
    }
#pragma unroll
    for (int p = 0; p < 2; ++p) {
      const int c = tid + p * 256;
      const int r = c >> 2, kc = (c & 3) * 8;
      *(uint4*)&As[buf][r][kc] = la[p];
      *(uint4*)&Bs[buf][r][kc] = lb[p];
    }
  };
  auto compute = [&](int buf) {
    v16bf af[2];
#pragma unroll
    for (int mt = 0; mt < 2; ++mt) {
      const int row = wr * 32 + mt * 16 + l;
      for (int i = 0; i < 16; ++i) af[mt][i] = As[buf][row][kidxA(i, half)];
    }
#pragma unroll
    for (int nt = 0; nt < 4; ++nt) {
      const int col = wc * 64 + nt * 16 + l;
      v16bf bfr;
      for (int i = 0; i < 16; ++i) bfr[i] = Bs[buf][col][16 * half + i];
      acc[0][nt] = wmma_bf16(af[0], bfr, acc[0][nt]);
      acc[1][nt] = wmma_bf16(af[1], bfr, acc[1][nt]);
    }
  };

  stage(0, 0);
  __syncthreads();

  int kt = 0;
  for (;;) {
    {  // phase 0: stage next slab into buf1, compute buf0; 1 barrier/slab
      if (kt + BKK < K) stage(kt + BKK, 1);
      compute(0);
      __syncthreads();
      kt += BKK;
      if (kt >= K) break;
    }
    {  // phase 1: stage next slab into buf0, compute buf1
      if (kt + BKK < K) stage(kt + BKK, 0);
      compute(1);
      __syncthreads();
      kt += BKK;
      if (kt >= K) break;
    }
  }

  // ---- epilogue (C layout: VGPR r -> M=r+8*half, N=lane&15)
  for (int mt = 0; mt < 2; ++mt)
    for (int nt = 0; nt < 4; ++nt)
      for (int r = 0; r < 8; ++r) {
        const int row = bm + wr * 32 + mt * 16 + 8 * half + r;
        const int col = bn + wc * 64 + nt * 16 + l;
        const size_t o = (size_t)row * N + col;
        float v = acc[mt][nt][r];
        if (mode == 2) {
          const float g = aux[o];
          Cb[o] = f2bf(v * (g / (1.f + __expf(-g))));
        } else {
          if (mode == 1) v += aux[o];
          Cf[o] = v;
        }
      }
}

// ---------------------------------------------------------------- RMSNorm
__global__ __launch_bounds__(256)
void k_rmsnorm(const float* __restrict__ x, const float* __restrict__ w,
               bf16* __restrict__ out) {
  const int row = blockIdx.x, tid = threadIdx.x;
  const float* xr = x + (size_t)row * DVAL;
  float s = 0.f;
  for (int j = tid; j < DVAL; j += 256) { const float v = xr[j]; s += v * v; }
  __shared__ float red[256];
  red[tid] = s; __syncthreads();
  for (int st = 128; st; st >>= 1) {
    if (tid < st) red[tid] += red[tid + st];
    __syncthreads();
  }
  const float rs = rsqrtf(red[0] * (1.f / DVAL) + 1e-6f);
  bf16* orow = out + (size_t)row * DVAL;
  for (int j = tid; j < DVAL; j += 256) orow[j] = f2bf(xr[j] * rs * w[j]);
}

// ---------------------------------------------------------------- RoPE + layout
// in : q,k,v fp32 [rows, H*DH] ; pos fp32 [rows] (null -> s index)
// out: qb,kb bf16 [B,H,S,DH] ; vt bf16 [B,H,DH,S]
__global__ __launch_bounds__(256)
void k_rope(const float* __restrict__ q, const float* __restrict__ k,
            const float* __restrict__ v, const float* __restrict__ pos,
            bf16* __restrict__ qb, bf16* __restrict__ kb, bf16* __restrict__ vt,
            int S) {
  const int i = blockIdx.x * 256 + threadIdx.x;   // rows*H*32 threads
  const int j = i & 31, h = (i >> 5) & (HVAL - 1), row = i >> 9;
  const int s = row % S, b = row / S;
  const float p = pos ? pos[row] : (float)s;
  const float inv = __expf(-(float)j * 0.28782314f);   // ln(10000)/32
  float sn, cs; __sincosf(p * inv, &sn, &cs);
  const size_t ib = (size_t)row * DVAL + h * DHV + j;
  const float q1 = q[ib], q2 = q[ib + 32];
  const float k1 = k[ib], k2 = k[ib + 32];
  const float v1 = v[ib], v2 = v[ib + 32];
  const size_t ob = ((size_t)(b * HVAL + h) * S + s) * DHV + j;
  qb[ob]      = f2bf(q1 * cs - q2 * sn);
  qb[ob + 32] = f2bf(q2 * cs + q1 * sn);
  kb[ob]      = f2bf(k1 * cs - k2 * sn);
  kb[ob + 32] = f2bf(k2 * cs + k1 * sn);
  const size_t vb = ((size_t)(b * HVAL + h) * DHV + j) * S + s;
  vt[vb]                  = f2bf(v1);
  vt[vb + (size_t)32 * S] = f2bf(v2);
}

// ---------------------------------------------------------------- attention
// flash-style, WMMA for QK^T and PV; wave = 16 queries, key blocks of 32
__global__ __launch_bounds__(256)
void k_attn(const bf16* __restrict__ qb, const bf16* __restrict__ kb,
            const bf16* __restrict__ vt, bf16* __restrict__ o, int S) {
  __shared__ bf16 P[8][16][TST];   // per-wave C->A relayout buffer
  const int tid = threadIdx.x, lane = tid & 31, w = tid >> 5;
  const int half = lane >> 4, l = lane & 15;
  const int h = blockIdx.y, b = blockIdx.z;
  const int q0 = blockIdx.x * 128 + w * 16;

  const bf16* qp = qb + (size_t)(b * HVAL + h) * S * DHV;
  const bf16* kp = kb + (size_t)(b * HVAL + h) * S * DHV;
  const bf16* vp = vt + (size_t)(b * HVAL + h) * DHV * S;

  v16bf aq[2];
  for (int f = 0; f < 2; ++f)
    for (int i = 0; i < 16; ++i)
      aq[f][i] = qp[(size_t)(q0 + l) * DHV + 32 * f + kidxA(i, half)];

  float m[8], ssum[8]; v8f oa[4];
  for (int r = 0; r < 8; ++r) { m[r] = -1e30f; ssum[r] = 0.f; }
  for (int t = 0; t < 4; ++t) oa[t] = vzero8();

  const float scale = 0.125f;  // 1/sqrt(64)
  const int kend = q0 + 16;    // causal: max key+1 for this wave
  for (int k0 = 0; k0 < kend; k0 += 32) {
    v8f sc[2]; sc[0] = vzero8(); sc[1] = vzero8();
    for (int kc = 0; kc < 2; ++kc)
      for (int n = 0; n < 2; ++n) {
        const bf16* base = kp + (size_t)(k0 + 16 * n + l) * DHV + 32 * kc + 16 * half;
        const v16bf bk = *(const v16bf*)base;
        sc[n] = wmma_bf16(aq[kc], bk, sc[n]);
      }
    for (int r = 0; r < 8; ++r) {
      const int qr = q0 + 8 * half + r;
      float s0 = sc[0][r] * scale, s1 = sc[1][r] * scale;
      if (k0 + l      > qr) s0 = -1e30f;
      if (k0 + 16 + l > qr) s1 = -1e30f;
      float lm = fmaxf(s0, s1);
      for (int msk = 1; msk < 16; msk <<= 1) lm = fmaxf(lm, __shfl_xor(lm, msk, 32));
      const float nm = fmaxf(m[r], lm);
      const float p0 = __expf(s0 - nm), p1 = __expf(s1 - nm);
      float ls = p0 + p1;
      for (int msk = 1; msk < 16; msk <<= 1) ls += __shfl_xor(ls, msk, 32);
      const float corr = __expf(m[r] - nm);
      ssum[r] = ssum[r] * corr + ls;
      m[r] = nm;
      for (int t = 0; t < 4; ++t) oa[t][r] *= corr;
      P[w][8 * half + r][l]      = f2bf(p0);
      P[w][8 * half + r][16 + l] = f2bf(p1);
    }
    v16bf ap;
    for (int i = 0; i < 16; ++i) ap[i] = P[w][l][kidxA(i, half)];
    for (int t = 0; t < 4; ++t) {
      const bf16* base = vp + (size_t)(16 * t + l) * S + k0 + 16 * half;
      const v16bf bv = *(const v16bf*)base;
      oa[t] = wmma_bf16(ap, bv, oa[t]);
    }
  }
  for (int r = 0; r < 8; ++r) {
    const int row = q0 + 8 * half + r;
    const float invs = 1.f / ssum[r];
    for (int t = 0; t < 4; ++t)
      o[(size_t)(b * S + row) * DVAL + h * DHV + 16 * t + l] =
          f2bf(oa[t][r] * invs);
  }
}

// ---------------------------------------------------------------- router
__global__ __launch_bounds__(256)
void k_router_stats(const float* __restrict__ proc, const float* __restrict__ xin,
                    const float* __restrict__ ml, float* __restrict__ d_st,
                    float* __restrict__ d_ch) {
  const int row = blockIdx.x, tid = threadIdx.x;
  const float* pr = proc + (size_t)row * DVAL;
  const float* xr = xin + (size_t)row * DVAL;
  const float* mu = ml + (size_t)row * 2 * DVAL;
  const float* lv = mu + DVAL;
  float s1 = 0.f, s2 = 0.f;
  for (int j = tid; j < DVAL; j += 256) {
    const float res = pr[j] - xr[j];
    s1 += res * res;
    const float lg = lv[j], dm = res - mu[j];
    // C_VAR = 1 -> log(C_VAR) = 0
    s2 += 0.5f * (lg + (1.0f + dm * dm) * __expf(-lg) - 1.0f);
  }
  __shared__ float r1[256], r2[256];
  r1[tid] = s1; r2[tid] = s2; __syncthreads();
  for (int st = 128; st; st >>= 1) {
    if (tid < st) { r1[tid] += r1[tid + st]; r2[tid] += r2[tid + st]; }
    __syncthreads();
  }
  if (tid == 0) {
    d_st[row] = r1[0] * (1.f / DVAL);
    d_ch[row] = r2[0] * (1.f / DVAL);
  }
}

__global__ __launch_bounds__(256)
void k_sums(const float* __restrict__ a, const float* __restrict__ b,
            float* __restrict__ out, int n) {
  const int tid = threadIdx.x;
  float s1 = 0.f, s2 = 0.f;
  for (int i = tid; i < n; i += 256) { s1 += a[i]; s2 += b[i]; }
  __shared__ float r1[256], r2[256];
  r1[tid] = s1; r2[tid] = s2; __syncthreads();
  for (int st = 128; st; st >>= 1) {
    if (tid < st) { r1[tid] += r1[tid + st]; r2[tid] += r2[tid + st]; }
    __syncthreads();
  }
  if (tid == 0) { out[0] = r1[0]; out[1] = r2[0]; }
}

__global__ __launch_bounds__(256)
void k_gate(const float* __restrict__ dst, const float* __restrict__ dch,
            const float* __restrict__ sums, float* __restrict__ g, int n) {
  const int i = blockIdx.x * 256 + threadIdx.x;
  const float m1 = sums[0] / (float)n, m2 = sums[1] / (float)n;
  const float z = (dst[i] - m1) + (dch[i] - m2);   // BETA_CE = BETA_CU = 1
  g[i] = 1.f / (1.f + __expf(-z));
}

// stable top-k (matches lax.top_k ties + argsort-by-index reordering)
__global__ __launch_bounds__(256)
void k_topk(const float* __restrict__ g, int* __restrict__ idx,
            float* __restrict__ sc, float* __restrict__ pos2) {
  __shared__ float gs[TVAL];
  __shared__ unsigned char fl[TVAL];
  const int b = blockIdx.x, tid = threadIdx.x;
  const float* gb = g + (size_t)b * TVAL;
  for (int i = tid; i < TVAL; i += 256) gs[i] = gb[i];
  __syncthreads();
  for (int i = tid; i < TVAL; i += 256) {
    const float gi = gs[i];
    int rank = 0;
    for (int j = 0; j < TVAL; ++j) {
      const float gj = gs[j];
      rank += (gj > gi) || (gj == gi && j < i);
    }
    fl[i] = (rank < KSEL) ? 1 : 0;
  }
  __syncthreads();
  for (int i = tid; i < TVAL; i += 256) {
    if (fl[i]) {
      int pos = 0;
      for (int j = 0; j < i; ++j) pos += fl[j];
      idx[b * KSEL + pos] = i;
      sc[b * KSEL + pos] = gs[i];
      pos2[b * KSEL + pos] = (float)i;
    }
  }
}

__global__ __launch_bounds__(256)
void k_gather(const float* __restrict__ proc, const int* __restrict__ idx,
              float* __restrict__ sel) {
  const size_t i = (size_t)blockIdx.x * 256 + threadIdx.x;
  const int d = i & (DVAL - 1);
  const int j = (i >> 10) & (KSEL - 1);
  const int b = (int)(i >> 18);
  sel[i] = proc[((size_t)b * TVAL + idx[b * KSEL + j]) * DVAL + d];
}

__global__ __launch_bounds__(256)
void k_copy(float* __restrict__ dst, const float* __restrict__ src, size_t n) {
  const size_t i = (size_t)blockIdx.x * 256 + threadIdx.x;
  if (i < n) dst[i] = src[i];
}

__global__ __launch_bounds__(256)
void k_scatter(float* __restrict__ out, const float* __restrict__ sel,
               const float* __restrict__ y, const float* __restrict__ sc,
               const int* __restrict__ idx) {
  const size_t i = (size_t)blockIdx.x * 256 + threadIdx.x;
  const int d = i & (DVAL - 1);
  const int j = (i >> 10) & (KSEL - 1);
  const int b = (int)(i >> 18);
  const float s = sel[i];
  out[((size_t)b * TVAL + idx[b * KSEL + j]) * DVAL + d] =
      s + sc[b * KSEL + j] * (y[i] - s);
}

// ---------------------------------------------------------------- host
extern "C" void kernel_launch(void* const* d_in, const int* in_sizes, int n_in,
                              void* d_out, int out_size, void* d_ws, size_t ws_size,
                              hipStream_t stream) {
  (void)in_sizes; (void)n_in; (void)out_size; (void)ws_size;
  const float* x     = (const float*)d_in[0];
  const float* wq1   = (const float*)d_in[1];
  const float* wk1   = (const float*)d_in[2];
  const float* wv1   = (const float*)d_in[3];
  const float* wo1   = (const float*)d_in[4];
  const float* wq2   = (const float*)d_in[5];
  const float* wk2   = (const float*)d_in[6];
  const float* wv2   = (const float*)d_in[7];
  const float* wo2   = (const float*)d_in[8];
  const float* wg1   = (const float*)d_in[9];
  const float* wu1   = (const float*)d_in[10];
  const float* wd1   = (const float*)d_in[11];
  const float* wg2   = (const float*)d_in[12];
  const float* wu2   = (const float*)d_in[13];
  const float* wd2   = (const float*)d_in[14];
  const float* ln1_1 = (const float*)d_in[15];
  const float* ln2_1 = (const float*)d_in[16];
  const float* ln1_2 = (const float*)d_in[17];
  const float* ln2_2 = (const float*)d_in[18];
  const float* pnw   = (const float*)d_in[19];
  const float* pw    = (const float*)d_in[20];

  char* base = (char*)d_ws;
  size_t off = 0;
  auto alloc = [&](size_t bytes) -> char* {
    char* r = base + off;
    off += (bytes + 255) & ~(size_t)255;
    return r;
  };

  const int rows1 = BVAL * TVAL;                     // 4096
  // bf16 transposed weights (converted once per launch)
  bf16* wqt[2]; bf16* wkt[2]; bf16* wvt[2]; bf16* wot[2];
  bf16* wgt[2]; bf16* wut[2]; bf16* wdt[2];
  for (int s = 0; s < 2; ++s) {
    wqt[s] = (bf16*)alloc((size_t)DVAL * DVAL * 2);
    wkt[s] = (bf16*)alloc((size_t)DVAL * DVAL * 2);
    wvt[s] = (bf16*)alloc((size_t)DVAL * DVAL * 2);
    wot[s] = (bf16*)alloc((size_t)DVAL * DVAL * 2);
    wgt[s] = (bf16*)alloc((size_t)DVAL * FFNV * 2);
    wut[s] = (bf16*)alloc((size_t)DVAL * FFNV * 2);
    wdt[s] = (bf16*)alloc((size_t)FFNV * DVAL * 2);
  }
  bf16* pwt = (bf16*)alloc((size_t)DVAL * 2 * DVAL * 2);

  float* processed = (float*)alloc((size_t)rows1 * DVAL * 4);
  float* yproc     = (float*)alloc((size_t)BVAL * KSEL * DVAL * 4);
  float* x2        = (float*)alloc((size_t)rows1 * DVAL * 4);
  bf16*  hbuf      = (bf16*)alloc((size_t)rows1 * DVAL * 2);
  float* big1      = (float*)alloc((size_t)rows1 * FFNV * 4);  // q|k|v, then FFN gate
  bf16*  gubuf     = (bf16*)alloc((size_t)rows1 * FFNV * 2);
  bf16*  obuf      = (bf16*)alloc((size_t)rows1 * DVAL * 2);
  bf16*  qbb       = (bf16*)alloc((size_t)rows1 * DVAL * 2);
  bf16*  kbb       = (bf16*)alloc((size_t)rows1 * DVAL * 2);
  bf16*  vtb       = (bf16*)alloc((size_t)rows1 * DVAL * 2);
  float* mlbuf     = (float*)alloc((size_t)rows1 * 2 * DVAL * 4);
  float* dstb      = (float*)alloc((size_t)rows1 * 4);
  float* dchb      = (float*)alloc((size_t)rows1 * 4);
  float* gbuf      = (float*)alloc((size_t)rows1 * 4);
  float* sums      = (float*)alloc(256);
  int*   idxb      = (int*)alloc((size_t)BVAL * KSEL * 4);
  float* scb       = (float*)alloc((size_t)BVAL * KSEL * 4);
  float* pos2b     = (float*)alloc((size_t)BVAL * KSEL * 4);
  float* selbuf    = (float*)alloc((size_t)BVAL * KSEL * DVAL * 4);

  auto cvt = [&](const float* src, bf16* dst, int Kd, int Nd) {
    k_cvt_t<<<dim3(Nd / 32, Kd / 32), 256, 0, stream>>>(src, dst, Kd, Nd);
  };
  cvt(wq1, wqt[0], DVAL, DVAL); cvt(wk1, wkt[0], DVAL, DVAL);
  cvt(wv1, wvt[0], DVAL, DVAL); cvt(wo1, wot[0], DVAL, DVAL);
  cvt(wq2, wqt[1], DVAL, DVAL); cvt(wk2, wkt[1], DVAL, DVAL);
  cvt(wv2, wvt[1], DVAL, DVAL); cvt(wo2, wot[1], DVAL, DVAL);
  cvt(wg1, wgt[0], DVAL, FFNV); cvt(wu1, wut[0], DVAL, FFNV);
  cvt(wd1, wdt[0], FFNV, DVAL);
  cvt(wg2, wgt[1], DVAL, FFNV); cvt(wu2, wut[1], DVAL, FFNV);
  cvt(wd2, wdt[1], FFNV, DVAL);
  cvt(pw, pwt, DVAL, 2 * DVAL);

  auto gemm = [&](const bf16* A, const bf16* Bt, float* Cf, bf16* Cb,
                  const float* aux, int M, int N, int K, int mode) {
    dim3 g(N / 128, M / 128);
    k_gemm<<<g, 256, 0, stream>>>(A, Bt, Cf, Cb, aux, M, N, K, mode);
  };

  auto decoder = [&](const float* xin, int S, const float* pos, int ws_id,
                     const float* ln1, const float* ln2, float* outp) {
    const int rows = BVAL * S;
    float* qf = big1;
    float* kf = big1 + (size_t)rows1 * DVAL;
    float* vf = big1 + (size_t)2 * rows1 * DVAL;
    k_rmsnorm<<<rows, 256, 0, stream>>>(xin, ln1, hbuf);
    gemm(hbuf, wqt[ws_id], qf, nullptr, nullptr, rows, DVAL, DVAL, 0);
    gemm(hbuf, wkt[ws_id], kf, nullptr, nullptr, rows, DVAL, DVAL, 0);
    gemm(hbuf, wvt[ws_id], vf, nullptr, nullptr, rows, DVAL, DVAL, 0);
    k_rope<<<rows * 2, 256, 0, stream>>>(qf, kf, vf, pos, qbb, kbb, vtb, S);
    k_attn<<<dim3(S / 128, HVAL, BVAL), 256, 0, stream>>>(qbb, kbb, vtb, obuf, S);
    gemm(obuf, wot[ws_id], x2, nullptr, xin, rows, DVAL, DVAL, 1);     // + residual
    k_rmsnorm<<<rows, 256, 0, stream>>>(x2, ln2, hbuf);
    gemm(hbuf, wgt[ws_id], big1, nullptr, nullptr, rows, FFNV, DVAL, 0);  // gate pre-act
    gemm(hbuf, wut[ws_id], nullptr, gubuf, big1, rows, FFNV, DVAL, 2);    // silu(g)*u -> bf16
    gemm(gubuf, wdt[ws_id], outp, nullptr, x2, rows, DVAL, FFNV, 1);      // down + residual
  };

  // decoder 1 on full sequence
  decoder(x, TVAL, nullptr, 0, ln1_1, ln2_1, processed);

  // prior + router
  k_rmsnorm<<<rows1, 256, 0, stream>>>(x, pnw, hbuf);
  gemm(hbuf, pwt, mlbuf, nullptr, nullptr, rows1, 2 * DVAL, DVAL, 0);
  k_router_stats<<<rows1, 256, 0, stream>>>(processed, x, mlbuf, dstb, dchb);
  k_sums<<<1, 256, 0, stream>>>(dstb, dchb, sums, rows1);
  k_gate<<<rows1 / 256, 256, 0, stream>>>(dstb, dchb, sums, gbuf, rows1);
  k_topk<<<BVAL, 256, 0, stream>>>(gbuf, idxb, scb, pos2b);
  k_gather<<<(BVAL * KSEL * DVAL) / 256, 256, 0, stream>>>(processed, idxb, selbuf);

  // decoder 2 on selected tokens (positions = original indices)
  decoder(selbuf, KSEL, pos2b, 1, ln1_2, ln2_2, yproc);

  // final: out = processed with selected rows blended
  k_copy<<<(unsigned)((size_t)rows1 * DVAL / 256), 256, 0, stream>>>(
      (float*)d_out, processed, (size_t)rows1 * DVAL);
  k_scatter<<<(BVAL * KSEL * DVAL) / 256, 256, 0, stream>>>(
      (float*)d_out, selbuf, yproc, scb, idxb);
}
